// AttentiveTransformer_53893249630225
// MI455X (gfx1250) — matrix operand order repaired and checked by reading, not compiled, not run on hardware
//
#include <hip/hip_runtime.h>
#include <hip/hip_bf16.h>

typedef __attribute__((ext_vector_type(16))) __bf16 v16bf;
typedef __attribute__((ext_vector_type(8)))  __bf16 v8bf;
typedef __attribute__((ext_vector_type(4)))  __bf16 v4bf;
typedef __attribute__((ext_vector_type(8)))  float  v8f;
typedef __attribute__((ext_vector_type(4)))  float  v4f;

#define FDIM   2048
#define BROWS  8192
#define BM     128
#define BN     128
#define BK     32
#define KTILES (FDIM / BK)
#define LDSTR  40   // padded row stride (elements): 20 dwords -> conflict-free mod 64 banks

#if __has_builtin(__builtin_amdgcn_sched_barrier)
#define SCHED_FENCE() __builtin_amdgcn_sched_barrier(0)
#else
#define SCHED_FENCE()
#endif

// ---------------------------------------------------------------------------
// Kernel 0: fold Dense bias + BatchNorm(inference) into per-feature scale/bias
// ---------------------------------------------------------------------------
__global__ void at_prep_kernel(const float* __restrict__ b,
                               const float* __restrict__ gamma,
                               const float* __restrict__ beta,
                               const float* __restrict__ mean,
                               const float* __restrict__ var,
                               float* __restrict__ inv,
                               float* __restrict__ bias2) {
    int i = blockIdx.x * blockDim.x + threadIdx.x;
    if (i < FDIM) {
        float iv = gamma[i] * rsqrtf(var[i] + 1e-3f);
        inv[i]   = iv;
        bias2[i] = b[i] * iv + beta[i] - mean[i] * iv;
    }
}

// ---------------------------------------------------------------------------
// Kernel 0b: pre-split A (fp32, row-major) into bf16 hi/lo arrays (row-major)
// ---------------------------------------------------------------------------
__global__ __launch_bounds__(256)
void at_convA_kernel(const float* __restrict__ A,
                     __bf16* __restrict__ hi, __bf16* __restrict__ lo) {
    size_t i = ((size_t)blockIdx.x * blockDim.x + threadIdx.x) * 4;
    v4f v = __builtin_nontemporal_load((const v4f*)(A + i));
    v4bf h, l;
#pragma unroll
    for (int j = 0; j < 4; ++j) {
        float  x = v[j];
        __bf16 hh = (__bf16)x;
        h[j] = hh;
        l[j] = (__bf16)(x - (float)hh);
    }
    *(v4bf*)(hi + i) = h;
    *(v4bf*)(lo + i) = l;
}

// ---------------------------------------------------------------------------
// Kernel 0c: pre-split W (fp32 [k][n]) into bf16 hi/lo TRANSPOSED arrays [n][k]
// 32x32 tiles through LDS so both global read and write are coalesced.
// ---------------------------------------------------------------------------
__global__ __launch_bounds__(256)
void at_convW_kernel(const float* __restrict__ W,
                     __bf16* __restrict__ hi, __bf16* __restrict__ lo) {
    __shared__ float tile[32][33];
    const int n0 = blockIdx.x * 32;
    const int k0 = blockIdx.y * 32;
    const int c  = threadIdx.x & 31;
    const int r  = threadIdx.x >> 5;   // 0..7
#pragma unroll
    for (int i = 0; i < 4; ++i) {
        int k = r + i * 8;
        tile[k][c] = __builtin_nontemporal_load(W + (size_t)(k0 + k) * FDIM + n0 + c);
    }
    __syncthreads();
#pragma unroll
    for (int i = 0; i < 4; ++i) {
        int    n = r + i * 8;
        float  x = tile[c][n];
        __bf16 h = (__bf16)x;
        size_t o = (size_t)(n0 + n) * FDIM + k0 + c;
        hi[o] = h;
        lo[o] = (__bf16)(x - (float)h);
    }
}

// ---------------------------------------------------------------------------
// Fast GEMM: operands pre-split to bf16 hi/lo (B pre-transposed to [n][k]).
// Double-buffered LDS; hot loop is pure b128 copy + ds frag loads + 24 WMMAs.
// Block tile 128x128, 8 waves (4 along M x 2 along N), wave tile 32x64.
// ---------------------------------------------------------------------------
__global__ __launch_bounds__(256, 1)
void at_gemm_fast_kernel(const __bf16* __restrict__ Ahi, const __bf16* __restrict__ Alo,
                         const __bf16* __restrict__ Bhi, const __bf16* __restrict__ Blo,
                         const float* __restrict__ inv, const float* __restrict__ bias2,
                         const float* __restrict__ prior, float* __restrict__ Z) {
    __shared__ __align__(16) __bf16 sA[2][2][BM][LDSTR];  // [buf][hi/lo][row][k]
    __shared__ __align__(16) __bf16 sB[2][2][BN][LDSTR];  // [buf][hi/lo][n][k]

    const int tid     = threadIdx.x;
    const int lane    = tid & 31;
    const int wave    = tid >> 5;
    const int hf      = lane >> 4;
    const int l16     = lane & 15;
    const int block_n = blockIdx.x * BN;
    const int block_m = blockIdx.y * BM;
    const int waveM   = (wave & 3) * 32;
    const int waveN   = (wave >> 2) * 64;

    // staging map: each thread copies rows r0 and r0+64 at k-offset ko (16B chunks)
    const int r0 = tid >> 2;           // 0..63
    const int ko = (tid & 3) * 8;      // 0,8,16,24
    const size_t aOff0 = (size_t)(block_m + r0) * FDIM + ko;
    const size_t aOff1 = aOff0 + (size_t)64 * FDIM;
    const size_t bOff0 = (size_t)(block_n + r0) * FDIM + ko;
    const size_t bOff1 = bOff0 + (size_t)64 * FDIM;

    v8f acc[2][4];
#pragma unroll
    for (int mi = 0; mi < 2; ++mi)
#pragma unroll
        for (int ni = 0; ni < 4; ++ni)
#pragma unroll
            for (int r = 0; r < 8; ++r) acc[mi][ni][r] = 0.0f;

    // prologue: fill buffer 0
    v8bf t[8];
    t[0] = *(const v8bf*)(Ahi + aOff0);
    t[1] = *(const v8bf*)(Ahi + aOff1);
    t[2] = *(const v8bf*)(Alo + aOff0);
    t[3] = *(const v8bf*)(Alo + aOff1);
    t[4] = *(const v8bf*)(Bhi + bOff0);
    t[5] = *(const v8bf*)(Bhi + bOff1);
    t[6] = *(const v8bf*)(Blo + bOff0);
    t[7] = *(const v8bf*)(Blo + bOff1);
    *(v8bf*)&sA[0][0][r0][ko]      = t[0];
    *(v8bf*)&sA[0][0][r0 + 64][ko] = t[1];
    *(v8bf*)&sA[0][1][r0][ko]      = t[2];
    *(v8bf*)&sA[0][1][r0 + 64][ko] = t[3];
    *(v8bf*)&sB[0][0][r0][ko]      = t[4];
    *(v8bf*)&sB[0][0][r0 + 64][ko] = t[5];
    *(v8bf*)&sB[0][1][r0][ko]      = t[6];
    *(v8bf*)&sB[0][1][r0 + 64][ko] = t[7];
    __syncthreads();

    for (int kt = 0; kt < KTILES; ++kt) {
        const int cur = kt & 1;
        const int nxt = cur ^ 1;
        const bool more = (kt + 1) < KTILES;

        // issue global loads for the next chunk before computing this one;
        // sched_barrier pins the load clause ahead of the WMMA block so the
        // full compute phase covers the L2 latency.
        v8bf tn[8];
        if (more) {
            const size_t k1 = (size_t)(kt + 1) * BK;
            tn[0] = *(const v8bf*)(Ahi + aOff0 + k1);
            tn[1] = *(const v8bf*)(Ahi + aOff1 + k1);
            tn[2] = *(const v8bf*)(Alo + aOff0 + k1);
            tn[3] = *(const v8bf*)(Alo + aOff1 + k1);
            tn[4] = *(const v8bf*)(Bhi + bOff0 + k1);
            tn[5] = *(const v8bf*)(Bhi + bOff1 + k1);
            tn[6] = *(const v8bf*)(Blo + bOff0 + k1);
            tn[7] = *(const v8bf*)(Blo + bOff1 + k1);
        }
        SCHED_FENCE();

        // ---- A fragments (ISA 16-bit A 16x32 layout) ----
        v16bf fah[2], fal[2];
#pragma unroll
        for (int mi = 0; mi < 2; ++mi) {
            int m  = waveM + mi * 16 + l16;
            int kb = hf * 8;
            v8bf h0 = *(const v8bf*)&sA[cur][0][m][kb];
            v8bf h1 = *(const v8bf*)&sA[cur][0][m][kb + 16];
            fah[mi] = __builtin_shufflevector(h0, h1, 0,1,2,3,4,5,6,7,8,9,10,11,12,13,14,15);
            v8bf l0 = *(const v8bf*)&sA[cur][1][m][kb];
            v8bf l1 = *(const v8bf*)&sA[cur][1][m][kb + 16];
            fal[mi] = __builtin_shufflevector(l0, l1, 0,1,2,3,4,5,6,7,8,9,10,11,12,13,14,15);
        }
        // ---- B fragments + 24 WMMAs ----
#pragma unroll
        for (int ni = 0; ni < 4; ++ni) {
            int n  = waveN + ni * 16 + l16;
            int kb = hf * 8;
            v8bf b0 = *(const v8bf*)&sB[cur][0][n][kb];
            v8bf b1 = *(const v8bf*)&sB[cur][0][n][kb + 16];
            v16bf fbh = __builtin_shufflevector(b0, b1, 0,1,2,3,4,5,6,7,8,9,10,11,12,13,14,15);
            v8bf c0 = *(const v8bf*)&sB[cur][1][n][kb];
            v8bf c1 = *(const v8bf*)&sB[cur][1][n][kb + 16];
            v16bf fbl = __builtin_shufflevector(c0, c1, 0,1,2,3,4,5,6,7,8,9,10,11,12,13,14,15);
#pragma unroll
            for (int mi = 0; mi < 2; ++mi) {
                v8f c = acc[mi][ni];
                c = __builtin_amdgcn_wmma_f32_16x16x32_bf16(false, fal[mi], false, fbh,
                                                            (short)0, c, false, false);
                c = __builtin_amdgcn_wmma_f32_16x16x32_bf16(false, fah[mi], false, fbl,
                                                            (short)0, c, false, false);
                c = __builtin_amdgcn_wmma_f32_16x16x32_bf16(false, fah[mi], false, fbh,
                                                            (short)0, c, false, false);
                acc[mi][ni] = c;
            }
        }

        if (more) {  // store next chunk into the other buffer (no conflict with cur reads)
            *(v8bf*)&sA[nxt][0][r0][ko]      = tn[0];
            *(v8bf*)&sA[nxt][0][r0 + 64][ko] = tn[1];
            *(v8bf*)&sA[nxt][1][r0][ko]      = tn[2];
            *(v8bf*)&sA[nxt][1][r0 + 64][ko] = tn[3];
            *(v8bf*)&sB[nxt][0][r0][ko]      = tn[4];
            *(v8bf*)&sB[nxt][0][r0 + 64][ko] = tn[5];
            *(v8bf*)&sB[nxt][1][r0][ko]      = tn[6];
            *(v8bf*)&sB[nxt][1][r0 + 64][ko] = tn[7];
        }
        __syncthreads();
    }

    // ---- epilogue: BN-affine + prior scale ----
#pragma unroll
    for (int ni = 0; ni < 4; ++ni) {
        int   col = block_n + waveN + ni * 16 + l16;
        float iv  = inv[col];
        float b2  = bias2[col];
#pragma unroll
        for (int mi = 0; mi < 2; ++mi) {
#pragma unroll
            for (int r = 0; r < 8; ++r) {
                int    row = block_m + waveM + mi * 16 + r + 8 * hf;
                size_t off = (size_t)row * FDIM + col;
                float  v   = acc[mi][ni][r] * iv + b2;
                Z[off] = v * __builtin_nontemporal_load(prior + off);
            }
        }
    }
}

// ---------------------------------------------------------------------------
// Fallback GEMM (round-1): fp32 operands, hi/lo split fused into staging.
// Used only if workspace is too small for the pre-split operand arrays.
// ---------------------------------------------------------------------------
__global__ __launch_bounds__(256, 1)
void at_gemm_kernel(const float* __restrict__ A,
                    const float* __restrict__ W,
                    const float* __restrict__ inv,
                    const float* __restrict__ bias2,
                    const float* __restrict__ prior,
                    float* __restrict__ Z) {
    __shared__ __align__(16) __bf16 sAhi[BM][LDSTR];
    __shared__ __align__(16) __bf16 sAlo[BM][LDSTR];
    __shared__ __align__(16) __bf16 sBhi[BN][LDSTR];
    __shared__ __align__(16) __bf16 sBlo[BN][LDSTR];

    const int tid     = threadIdx.x;
    const int lane    = tid & 31;
    const int wave    = tid >> 5;
    const int hf      = lane >> 4;
    const int l16     = lane & 15;
    const int block_n = blockIdx.x * BN;
    const int block_m = blockIdx.y * BM;
    const int waveM   = (wave & 3) * 32;
    const int waveN   = (wave >> 2) * 64;

    v8f acc[2][4];
#pragma unroll
    for (int mi = 0; mi < 2; ++mi)
#pragma unroll
        for (int ni = 0; ni < 4; ++ni)
#pragma unroll
            for (int r = 0; r < 8; ++r) acc[mi][ni][r] = 0.0f;

    for (int k0 = 0; k0 < FDIM; k0 += BK) {
#pragma unroll
        for (int p = 0; p < 4; ++p) {
            int r = (tid >> 3) + p * 32;
            int c = (tid & 7) * 4;
            v4f v = *(const v4f*)(A + (size_t)(block_m + r) * FDIM + k0 + c);
#pragma unroll
            for (int i = 0; i < 4; ++i) {
                float  x = v[i];
                __bf16 h = (__bf16)x;
                sAhi[r][c + i] = h;
                sAlo[r][c + i] = (__bf16)(x - (float)h);
            }
        }
#pragma unroll
        for (int p = 0; p < 4; ++p) {
            int k = (tid >> 5) + p * 8;
            int c = (tid & 31) * 4;
            v4f v = *(const v4f*)(W + (size_t)(k0 + k) * FDIM + block_n + c);
#pragma unroll
            for (int i = 0; i < 4; ++i) {
                float  x = v[i];
                __bf16 h = (__bf16)x;
                sBhi[c + i][k] = h;
                sBlo[c + i][k] = (__bf16)(x - (float)h);
            }
        }
        __syncthreads();

        v16bf ahi[2], alo[2];
#pragma unroll
        for (int mi = 0; mi < 2; ++mi) {
            int m  = waveM + mi * 16 + l16;
            int kb = hf * 8;
            v8bf h0 = *(const v8bf*)&sAhi[m][kb];
            v8bf h1 = *(const v8bf*)&sAhi[m][kb + 16];
            ahi[mi] = __builtin_shufflevector(h0, h1, 0,1,2,3,4,5,6,7,8,9,10,11,12,13,14,15);
            v8bf l0 = *(const v8bf*)&sAlo[m][kb];
            v8bf l1 = *(const v8bf*)&sAlo[m][kb + 16];
            alo[mi] = __builtin_shufflevector(l0, l1, 0,1,2,3,4,5,6,7,8,9,10,11,12,13,14,15);
        }
#pragma unroll
        for (int ni = 0; ni < 4; ++ni) {
            int n  = waveN + ni * 16 + l16;
            int kb = hf * 8;
            v8bf b0 = *(const v8bf*)&sBhi[n][kb];
            v8bf b1 = *(const v8bf*)&sBhi[n][kb + 16];
            v16bf bhi = __builtin_shufflevector(b0, b1, 0,1,2,3,4,5,6,7,8,9,10,11,12,13,14,15);
            v8bf c0 = *(const v8bf*)&sBlo[n][kb];
            v8bf c1 = *(const v8bf*)&sBlo[n][kb + 16];
            v16bf blo = __builtin_shufflevector(c0, c1, 0,1,2,3,4,5,6,7,8,9,10,11,12,13,14,15);
#pragma unroll
            for (int mi = 0; mi < 2; ++mi) {
                v8f c = acc[mi][ni];
                c = __builtin_amdgcn_wmma_f32_16x16x32_bf16(false, alo[mi], false, bhi,
                                                            (short)0, c, false, false);
                c = __builtin_amdgcn_wmma_f32_16x16x32_bf16(false, ahi[mi], false, blo,
                                                            (short)0, c, false, false);
                c = __builtin_amdgcn_wmma_f32_16x16x32_bf16(false, ahi[mi], false, bhi,
                                                            (short)0, c, false, false);
                acc[mi][ni] = c;
            }
        }
        __syncthreads();
    }

#pragma unroll
    for (int ni = 0; ni < 4; ++ni) {
        int   col = block_n + waveN + ni * 16 + l16;
        float iv  = inv[col];
        float b2  = bias2[col];
#pragma unroll
        for (int mi = 0; mi < 2; ++mi) {
#pragma unroll
            for (int r = 0; r < 8; ++r) {
                int    row = block_m + waveM + mi * 16 + r + 8 * hf;
                size_t off = (size_t)row * FDIM + col;
                float  v   = acc[mi][ni][r] * iv + b2;
                Z[off] = v * prior[off];
            }
        }
    }
}

// ---------------------------------------------------------------------------
// Kernel 2: in-place row sparsemax via bisection on tau in [max-1, max].
// ---------------------------------------------------------------------------
__device__ __forceinline__ float block_reduce(float v, float* red, int lane, int wave,
                                              bool is_max) {
#pragma unroll
    for (int off = 16; off >= 1; off >>= 1) {
        float o = __shfl_xor(v, off, 32);
        v = is_max ? fmaxf(v, o) : (v + o);
    }
    if (lane == 0) red[wave] = v;
    __syncthreads();
    float r = red[0];
#pragma unroll
    for (int w = 1; w < 8; ++w) r = is_max ? fmaxf(r, red[w]) : (r + red[w]);
    return r;
}

__global__ __launch_bounds__(256, 1)
void at_sparsemax_kernel(float* __restrict__ Z) {
    __shared__ float red[8];
    const int tid  = threadIdx.x;
    const int lane = tid & 31;
    const int wave = tid >> 5;
    float* z = Z + (size_t)blockIdx.x * FDIM;

    float v[8];
#pragma unroll
    for (int i = 0; i < 8; ++i) v[i] = z[tid + i * 256];

    float m = v[0];
#pragma unroll
    for (int i = 1; i < 8; ++i) m = fmaxf(m, v[i]);
    float rmax = block_reduce(m, red, lane, wave, true);

    float lo = rmax - 1.0f, hi = rmax;   // f(lo) >= 1 >= f(hi)
    for (int it = 0; it < 30; ++it) {
        float tau = 0.5f * (lo + hi);
        float s = 0.0f;
#pragma unroll
        for (int i = 0; i < 8; ++i) s += fmaxf(v[i] - tau, 0.0f);
        __syncthreads();
        s = block_reduce(s, red, lane, wave, false);
        if (s >= 1.0f) lo = tau; else hi = tau;
    }
    float tau = 0.5f * (lo + hi);
#pragma unroll
    for (int i = 0; i < 8; ++i) z[tid + i * 256] = fmaxf(v[i] - tau, 0.0f);
}

// ---------------------------------------------------------------------------
extern "C" void kernel_launch(void* const* d_in, const int* in_sizes, int n_in,
                              void* d_out, int out_size, void* d_ws, size_t ws_size,
                              hipStream_t stream) {
    const float* inputs = (const float*)d_in[0];
    const float* W      = (const float*)d_in[1];
    const float* b      = (const float*)d_in[2];
    const float* gamma  = (const float*)d_in[3];
    const float* beta   = (const float*)d_in[4];
    const float* mmean  = (const float*)d_in[5];
    const float* mvar   = (const float*)d_in[6];
    const float* prior  = (const float*)d_in[7];
    float* out = (float*)d_out;

    float*  inv   = (float*)d_ws;                 // [FDIM]
    float*  bias2 = inv + FDIM;                   // [FDIM]
    __bf16* Ahi   = (__bf16*)(bias2 + FDIM);
    __bf16* Alo   = Ahi + (size_t)BROWS * FDIM;
    __bf16* Bthi  = Alo + (size_t)BROWS * FDIM;   // W transposed [n][k]
    __bf16* Btlo  = Bthi + (size_t)FDIM * FDIM;

    const size_t need = 2u * FDIM * sizeof(float) +
                        (2ull * BROWS * FDIM + 2ull * FDIM * FDIM) * sizeof(__bf16);

    at_prep_kernel<<<(FDIM + 255) / 256, 256, 0, stream>>>(b, gamma, beta, mmean, mvar,
                                                           inv, bias2);

    dim3 grid(FDIM / BN, BROWS / BM);             // (16, 64)
    if (ws_size >= need) {
        at_convA_kernel<<<(int)((size_t)BROWS * FDIM / 4 / 256), 256, 0, stream>>>(inputs, Ahi, Alo);
        at_convW_kernel<<<dim3(FDIM / 32, FDIM / 32), 256, 0, stream>>>(W, Bthi, Btlo);
        at_gemm_fast_kernel<<<grid, 256, 0, stream>>>(Ahi, Alo, Bthi, Btlo,
                                                      inv, bias2, prior, out);
    } else {
        at_gemm_kernel<<<grid, 256, 0, stream>>>(inputs, W, inv, bias2, prior, out);
    }

    at_sparsemax_kernel<<<BROWS, 256, 0, stream>>>(out);
}
